// InteractionBlock_Dropout_944892805681
// MI455X (gfx1250) — compile-verified, matrix operand order checked by reading, not run on hardware
//
#include <hip/hip_runtime.h>
#include <hip/hip_bf16.h>

// ---------------------------------------------------------------------------
// DimeNet++ interaction block for MI455X (gfx1250), wave32 + WMMA bf16.
// ---------------------------------------------------------------------------

typedef __attribute__((ext_vector_type(8)))  __bf16 v8bf;
typedef __attribute__((ext_vector_type(16))) __bf16 v16bf;
typedef __attribute__((ext_vector_type(8)))  float  v8f;

#define NE  262144    // edges
#define NT  2097152   // triplets
#define PACT 136      // LDS bf16 activation pitch -> 272B rows, bank skew
#define PFB  132      // LDS f32 staging pitch (edge_back) -> 528B rows
#define PFH  68       // LDS f32 staging pitch (edge_front h) -> 272B rows

static __device__ __forceinline__ float swish_f(float x) {
    return x * (1.0f / (1.0f + __expf(-x)));
}

static __device__ __forceinline__ v8f vzero8() {
    v8f a;
#pragma unroll
    for (int i = 0; i < 8; ++i) a[i] = 0.0f;
    return a;
}

static __device__ __forceinline__ v16bf mk16(v8bf lo, v8bf hi) {
    union { v16bf v; v8bf h[2]; } u;
    u.h[0] = lo; u.h[1] = hi;
    return u.v;
}

static __device__ __forceinline__ v8bf cvt8(const float* __restrict__ p) {
    v8bf r;
#pragma unroll
    for (int i = 0; i < 8; ++i) r[i] = (__bf16)p[i];
    return r;
}

static __device__ __forceinline__ v8f wmma_bf16(v16bf A, v16bf B, v8f C) {
    return __builtin_amdgcn_wmma_f32_16x16x32_bf16(
        false, A, false, B, (short)0, C, false, false);
}

// A fragment (16x32 bf16) from LDS activation tile
static __device__ __forceinline__ v16bf ldA(const __bf16* sAw, int ln, int hl,
                                            int kc) {
    const __bf16* ap = sAw + ln * PACT + kc * 32 + hl * 8;
    return mk16(*(const v8bf*)ap, *(const v8bf*)(ap + 16));
}

// B fragment (32x16 bf16) from col-major global weights Wt[N][K]
static __device__ __forceinline__ v16bf ldB(const __bf16* __restrict__ Wt, int K,
                                            int ncol, int hl, int kc) {
    const __bf16* bp = Wt + ncol * K + kc * 32 + hl * 8;
    return mk16(*(const v8bf*)bp, *(const v8bf*)(bp + 16));
}

// A fragment from contiguous f32 (converted)
static __device__ __forceinline__ v16bf ldA_g(const float* __restrict__ p) {
    return mk16(cvt8(p), cvt8(p + 16));
}

// acc[8][8] = sA(16x128) @ Wt(128x128); A-frags hoisted once
static __device__ __forceinline__ void gemm_lds(const __bf16* sAw,
                                                const __bf16* __restrict__ Wt,
                                                float acc[8][8], int ln, int hl) {
    v16bf Af[4];
#pragma unroll
    for (int kc = 0; kc < 4; ++kc) Af[kc] = ldA(sAw, ln, hl, kc);
#pragma unroll
    for (int jn = 0; jn < 8; ++jn) {
        v8f a = vzero8();
#pragma unroll
        for (int kc = 0; kc < 4; ++kc)
            a = wmma_bf16(Af[kc], ldB(Wt, 128, jn * 16 + ln, hl, kc), a);
#pragma unroll
        for (int r = 0; r < 8; ++r) acc[jn][r] = a[r];
    }
}

// x <- x + swish(swish(x@W0+b0)@W1+b1); sAw kept in sync (bf16)
static __device__ __forceinline__ void res_block(__bf16* sAw, float x[8][8],
                                                 const __bf16* __restrict__ W0,
                                                 const float* __restrict__ b0,
                                                 const __bf16* __restrict__ W1,
                                                 const float* __restrict__ b1,
                                                 int ln, int hl) {
    float t[8][8];
    gemm_lds(sAw, W0, t, ln, hl);
#pragma unroll
    for (int jn = 0; jn < 8; ++jn) {
        int col = jn * 16 + ln;
        float bb = b0[col];
#pragma unroll
        for (int r = 0; r < 8; ++r) {
            float v = swish_f(t[jn][r] + bb);
            t[jn][r] = v;
            sAw[(r + hl * 8) * PACT + col] = (__bf16)v;
        }
    }
    gemm_lds(sAw, W1, t, ln, hl);
#pragma unroll
    for (int jn = 0; jn < 8; ++jn) {
        int col = jn * 16 + ln;
        float bb = b1[col];
#pragma unroll
        for (int r = 0; r < 8; ++r) {
            float v = x[jn][r] + swish_f(t[jn][r] + bb);
            x[jn][r] = v;
            sAw[(r + hl * 8) * PACT + col] = (__bf16)v;
        }
    }
}

// ---------------------------------------------------------------------------
// prep: W[K][N] f32 (row-major) -> Wt[N][K] bf16 (col-major for B fragments)
// ---------------------------------------------------------------------------
__global__ void convT_kernel(const float* __restrict__ W, __bf16* __restrict__ Wt,
                             int K, int N) {
    int i = blockIdx.x * 256 + threadIdx.x;
    if (i >= K * N) return;
    int k = i / N, n = i % N;
    Wt[n * K + k] = (__bf16)W[i];
}

__global__ void zero_kernel(float4* __restrict__ p, int n4) {
    int i = blockIdx.x * 256 + threadIdx.x;
    if (i < n4) p[i] = make_float4(0.f, 0.f, 0.f, 0.f);
}

// ---------------------------------------------------------------------------
// edge_front: h = swish( (swish(m@W_kj+b) * rbf_gate) @ W_down )   [E, 64]
// ---------------------------------------------------------------------------
union ShFront {
    __bf16 a[8][16 * PACT];   // bf16 activation tiles (one per wave)
    float  f[8][16 * PFH];    // f32 h staging (same footprint, reused after)
};

__global__ __launch_bounds__(256) void edge_front_kernel(
    const float* __restrict__ m_input, const float* __restrict__ rbf,
    const float* __restrict__ W_rbf1, const float* __restrict__ W_rbf2,
    const float* __restrict__ b_kj,
    const __bf16* __restrict__ Wt_kj, const __bf16* __restrict__ Wt_down,
    float* __restrict__ h) {
    __shared__ ShFront sh;
    __shared__ float t1[128][8];

    const int tid = threadIdx.x;
    const int w = tid >> 5, L = tid & 31, hl = L >> 4, ln = L & 15;
    const int row0 = blockIdx.x * 128;
    const int rbase = row0 + w * 16;

    // t1[row][j] = rbf[row] @ W_rbf1  (128 rows x 8)
#pragma unroll
    for (int e = 0; e < 4; ++e) {
        int idx = tid * 4 + e;
        int rr = idx >> 3, j = idx & 7;
        const float* rp = rbf + (size_t)(row0 + rr) * 6;
        float s = 0.f;
#pragma unroll
        for (int k = 0; k < 6; ++k) s += rp[k] * W_rbf1[k * 8 + j];
        t1[rr][j] = s;
    }
    __syncthreads();

    __bf16* sAw = sh.a[w];

    // GEMM 1: swish(m_input @ W_kj + b) * rbf_gate -> LDS bf16
    v16bf Am[4];
#pragma unroll
    for (int kc = 0; kc < 4; ++kc)
        Am[kc] = ldA_g(m_input + (size_t)(rbase + ln) * 128 + kc * 32 + hl * 8);
#pragma unroll
    for (int jn = 0; jn < 8; ++jn) {
        v8f a = vzero8();
#pragma unroll
        for (int kc = 0; kc < 4; ++kc)
            a = wmma_bf16(Am[kc], ldB(Wt_kj, 128, jn * 16 + ln, hl, kc), a);
        int col = jn * 16 + ln;
        float bb = b_kj[col];
#pragma unroll
        for (int r = 0; r < 8; ++r) {
            int rl = r + hl * 8;
            float v = swish_f(a[r] + bb);
            const float* tp = t1[w * 16 + rl];
            float g = 0.f;
#pragma unroll
            for (int j = 0; j < 8; ++j) g += tp[j] * W_rbf2[j * 128 + col];
            sAw[rl * PACT + col] = (__bf16)(v * g);
        }
    }

    // GEMM 2: @ W_down (K=128, N=64), swish -> registers
    float hv[4][8];
    {
        v16bf Ad[4];
#pragma unroll
        for (int kc = 0; kc < 4; ++kc) Ad[kc] = ldA(sAw, ln, hl, kc);
#pragma unroll
        for (int jn = 0; jn < 4; ++jn) {
            v8f a = vzero8();
#pragma unroll
            for (int kc = 0; kc < 4; ++kc)
                a = wmma_bf16(Ad[kc], ldB(Wt_down, 128, jn * 16 + ln, hl, kc), a);
#pragma unroll
            for (int r = 0; r < 8; ++r) hv[jn][r] = swish_f(a[r]);
        }
    }

    // stage to LDS (f32) and store coalesced
    __syncthreads();   // all waves done reading sh.a before aliased f32 reuse
    float* sFw = sh.f[w];
#pragma unroll
    for (int jn = 0; jn < 4; ++jn) {
        int col = jn * 16 + ln;
#pragma unroll
        for (int r = 0; r < 8; ++r) sFw[(r + hl * 8) * PFH + col] = hv[jn][r];
    }
#pragma unroll
    for (int row = 0; row < 16; ++row) {
        float2 v = *(const float2*)&sFw[row * PFH + L * 2];
        *(float2*)&h[(size_t)(rbase + row) * 64 + L * 2] = v;
    }
}

// ---------------------------------------------------------------------------
// triplet: agg[reduce] += h[expand] * ((sbf@W_sbf1)@W_sbf2)
// ---------------------------------------------------------------------------
__global__ __launch_bounds__(256) void triplet_kernel(
    const float* __restrict__ sbf, const int* __restrict__ idx_kj,
    const int* __restrict__ idx_ji, const float* __restrict__ W_sbf1,
    const float* __restrict__ W_sbf2, const float* __restrict__ h,
    float* __restrict__ agg) {
    __shared__ float ssbf[64][44];
    __shared__ float st1[64][8];
    __shared__ float sW1[336];
    __shared__ float sW2[512];

    const int tid = threadIdx.x;
    const long t0 = (long)blockIdx.x * 64;

    // prefetch next block's sbf stream (global_prefetch_b8)
    if (blockIdx.x + 1 < gridDim.x) {
        const char* nxt = (const char*)(sbf + (t0 + 64) * 42);
        for (int i = tid; i < 84; i += 256)
            __builtin_prefetch(nxt + (size_t)i * 128, 0, 1);
    }

    if (tid < 336) sW1[tid] = W_sbf1[tid];
    for (int i = tid; i < 512; i += 256) sW2[i] = W_sbf2[i];
    for (int i = tid; i < 64 * 42; i += 256)
        ssbf[i / 42][i % 42] = sbf[t0 * 42 + i];
    __syncthreads();

    // st1 = sbf_tile @ W_sbf1  (64 x 8)
    for (int i = tid; i < 512; i += 256) {
        int tt = i >> 3, j = i & 7;
        float s = 0.f;
#pragma unroll
        for (int k = 0; k < 42; ++k) s += ssbf[tt][k] * sW1[k * 8 + j];
        st1[tt][j] = s;
    }
    __syncthreads();

    const int lt = tid >> 2, cg = tid & 3;        // 4 threads per triplet
    const long t = t0 + lt;
    const int ik = idx_kj[t], ij = idx_ji[t];
    const float* hp = h + (size_t)ik * 64 + cg * 16;
    float* ap = agg + (size_t)ij * 64 + cg * 16;

    float tl[8];
#pragma unroll
    for (int j = 0; j < 8; ++j) tl[j] = st1[lt][j];

#pragma unroll
    for (int c = 0; c < 16; ++c) {
        float g = 0.f;
#pragma unroll
        for (int j = 0; j < 8; ++j) g += tl[j] * sW2[j * 64 + cg * 16 + c];
        atomicAdd(&ap[c], hp[c] * g);
    }
}

// ---------------------------------------------------------------------------
// edge_back: m = ResA2( swish(ResB(swish(m@W_ji+b)+swish(agg@W_up))@W_f+b) + m )
// ---------------------------------------------------------------------------
union ShBack {
    __bf16 a[8][16 * PACT];   // bf16 activation tiles
    float  f[8][16 * PFB];    // f32 output staging (reused after last GEMM)
};

__global__ __launch_bounds__(256) void edge_back_kernel(
    const float* __restrict__ m_input, const float* __restrict__ agg,
    const __bf16* __restrict__ Wt_ji, const __bf16* __restrict__ Wt_up,
    const __bf16* __restrict__ Wt_rb0, const __bf16* __restrict__ Wt_rb1,
    const __bf16* __restrict__ Wt_fin,
    const __bf16* __restrict__ Wt_ra00, const __bf16* __restrict__ Wt_ra01,
    const __bf16* __restrict__ Wt_ra10, const __bf16* __restrict__ Wt_ra11,
    const float* __restrict__ b_ji, const float* __restrict__ b_rb,
    const float* __restrict__ b_fin, const float* __restrict__ b_ra,
    float* __restrict__ out) {
    __shared__ ShBack sh;

    const int tid = threadIdx.x;
    const int w = tid >> 5, L = tid & 31, hl = L >> 4, ln = L & 15;
    const int row0 = blockIdx.x * 128;
    const int rbase = row0 + w * 16;
    __bf16* sAw = sh.a[w];

    float x[8][8];

    // Step 1: x = swish(m@W_ji + b_ji) + swish(agg@W_up)
    {
        v16bf Am[4];
#pragma unroll
        for (int kc = 0; kc < 4; ++kc)
            Am[kc] = ldA_g(m_input + (size_t)(rbase + ln) * 128 + kc * 32 + hl * 8);
        v16bf Ag[2];
#pragma unroll
        for (int kc = 0; kc < 2; ++kc)
            Ag[kc] = ldA_g(agg + (size_t)(rbase + ln) * 64 + kc * 32 + hl * 8);
#pragma unroll
        for (int jn = 0; jn < 8; ++jn) {
            v8f a1 = vzero8();
#pragma unroll
            for (int kc = 0; kc < 4; ++kc)
                a1 = wmma_bf16(Am[kc], ldB(Wt_ji, 128, jn * 16 + ln, hl, kc), a1);
            v8f a2 = vzero8();
#pragma unroll
            for (int kc = 0; kc < 2; ++kc)
                a2 = wmma_bf16(Ag[kc], ldB(Wt_up, 64, jn * 16 + ln, hl, kc), a2);
            int col = jn * 16 + ln;
            float bj = b_ji[col];
#pragma unroll
            for (int r = 0; r < 8; ++r) {
                float v = swish_f(a1[r] + bj) + swish_f(a2[r]);
                x[jn][r] = v;
                sAw[(r + hl * 8) * PACT + col] = (__bf16)v;
            }
        }
    }

    // Residual block (before skip)
    res_block(sAw, x, Wt_rb0, b_rb, Wt_rb1, b_rb + 128, ln, hl);

    // Final dense + input skip
    {
        float t[8][8];
        gemm_lds(sAw, Wt_fin, t, ln, hl);
#pragma unroll
        for (int jn = 0; jn < 8; ++jn) {
            int col = jn * 16 + ln;
            float bb = b_fin[col];
#pragma unroll
            for (int r = 0; r < 8; ++r) {
                int rl = r + hl * 8;
                float v = swish_f(t[jn][r] + bb) +
                          m_input[(size_t)(rbase + rl) * 128 + col];
                x[jn][r] = v;
                sAw[rl * PACT + col] = (__bf16)v;
            }
        }
    }

    // Residual blocks (after skip)
    res_block(sAw, x, Wt_ra00, b_ra,       Wt_ra01, b_ra + 128, ln, hl);
    res_block(sAw, x, Wt_ra10, b_ra + 256, Wt_ra11, b_ra + 384, ln, hl);

    // Stage f32 result in LDS (aliases sh.a, dead now) and store coalesced
    __syncthreads();
    float* sFw = sh.f[w];
#pragma unroll
    for (int jn = 0; jn < 8; ++jn) {
        int col = jn * 16 + ln;
#pragma unroll
        for (int r = 0; r < 8; ++r) sFw[(r + hl * 8) * PFB + col] = x[jn][r];
    }
#pragma unroll
    for (int row = 0; row < 16; ++row) {
        float4 v = *(const float4*)&sFw[row * PFB + L * 4];
        *(float4*)&out[(size_t)(rbase + row) * 128 + L * 4] = v;
    }
}

// ---------------------------------------------------------------------------
extern "C" void kernel_launch(void* const* d_in, const int* in_sizes, int n_in,
                              void* d_out, int out_size, void* d_ws, size_t ws_size,
                              hipStream_t stream) {
    (void)in_sizes; (void)n_in; (void)out_size; (void)ws_size;

    const float* m_input = (const float*)d_in[0];
    const float* rbf     = (const float*)d_in[1];
    const float* sbf     = (const float*)d_in[2];
    const int*   e_kj    = (const int*)d_in[3];
    const int*   e_ji    = (const int*)d_in[4];
    const float* W_kj    = (const float*)d_in[5];
    const float* b_kj    = (const float*)d_in[6];
    const float* W_rbf1  = (const float*)d_in[7];
    const float* W_rbf2  = (const float*)d_in[8];
    const float* W_sbf1  = (const float*)d_in[9];
    const float* W_sbf2  = (const float*)d_in[10];
    const float* W_down  = (const float*)d_in[11];
    const float* W_up    = (const float*)d_in[12];
    const float* W_ji    = (const float*)d_in[13];
    const float* b_ji    = (const float*)d_in[14];
    const float* W_res_b = (const float*)d_in[15];
    const float* b_res_b = (const float*)d_in[16];
    const float* W_final = (const float*)d_in[17];
    const float* b_final = (const float*)d_in[18];
    const float* W_res_a = (const float*)d_in[19];
    const float* b_res_a = (const float*)d_in[20];

    // workspace layout
    char* ws = (char*)d_ws;
    float* h   = (float*)ws;                                   // E x 64 f32
    float* agg = (float*)(ws + (size_t)NE * 64 * 4);           // E x 64 f32
    __bf16* wb = (__bf16*)(ws + (size_t)NE * 64 * 4 * 2);      // bf16 weights
    __bf16* Wt_kj   = wb;               wb += 128 * 128;
    __bf16* Wt_ji   = wb;               wb += 128 * 128;
    __bf16* Wt_down = wb;               wb += 128 * 64;
    __bf16* Wt_up   = wb;               wb += 64 * 128;
    __bf16* Wt_rb0  = wb;               wb += 128 * 128;
    __bf16* Wt_rb1  = wb;               wb += 128 * 128;
    __bf16* Wt_fin  = wb;               wb += 128 * 128;
    __bf16* Wt_ra00 = wb;               wb += 128 * 128;
    __bf16* Wt_ra01 = wb;               wb += 128 * 128;
    __bf16* Wt_ra10 = wb;               wb += 128 * 128;
    __bf16* Wt_ra11 = wb;               wb += 128 * 128;

    // weight prep (transpose + bf16 convert)
    convT_kernel<<<64, 256, 0, stream>>>(W_kj,              Wt_kj,   128, 128);
    convT_kernel<<<64, 256, 0, stream>>>(W_ji,              Wt_ji,   128, 128);
    convT_kernel<<<32, 256, 0, stream>>>(W_down,            Wt_down, 128, 64);
    convT_kernel<<<32, 256, 0, stream>>>(W_up,              Wt_up,   64,  128);
    convT_kernel<<<64, 256, 0, stream>>>(W_res_b,           Wt_rb0,  128, 128);
    convT_kernel<<<64, 256, 0, stream>>>(W_res_b + 16384,   Wt_rb1,  128, 128);
    convT_kernel<<<64, 256, 0, stream>>>(W_final,           Wt_fin,  128, 128);
    convT_kernel<<<64, 256, 0, stream>>>(W_res_a,           Wt_ra00, 128, 128);
    convT_kernel<<<64, 256, 0, stream>>>(W_res_a + 16384,   Wt_ra01, 128, 128);
    convT_kernel<<<64, 256, 0, stream>>>(W_res_a + 32768,   Wt_ra10, 128, 128);
    convT_kernel<<<64, 256, 0, stream>>>(W_res_a + 49152,   Wt_ra11, 128, 128);

    // zero the segment-sum accumulator
    zero_kernel<<<(NE * 64 / 4 + 255) / 256, 256, 0, stream>>>((float4*)agg,
                                                               NE * 64 / 4);

    // stage 1: edge front-end (2 WMMA GEMMs fused)
    edge_front_kernel<<<NE / 128, 256, 0, stream>>>(m_input, rbf, W_rbf1, W_rbf2,
                                                    b_kj, Wt_kj, Wt_down, h);

    // stage 2: triplet gather/gate/scatter-add
    triplet_kernel<<<NT / 64, 256, 0, stream>>>(sbf, e_kj, e_ji, W_sbf1, W_sbf2,
                                                h, agg);

    // stage 3: edge back-end (9 WMMA GEMMs fused)
    edge_back_kernel<<<NE / 128, 256, 0, stream>>>(
        m_input, agg, Wt_ji, Wt_up, Wt_rb0, Wt_rb1, Wt_fin,
        Wt_ra00, Wt_ra01, Wt_ra10, Wt_ra11,
        b_ji, b_res_b, b_final, b_res_a, (float*)d_out);
}
